// DirectionalScan_41669772706452
// MI455X (gfx1250) — matrix operand order repaired and checked by reading, not compiled, not run on hardware
//
#include <hip/hip_runtime.h>
#include <hip/hip_bf16.h>

// ---- problem dims (fixed by reference) ----
#define BATCH   4
#define HDIM    64
#define WDIM    64
#define DMODEL  512
#define NSTATE  8
#define LTOT    (HDIM * WDIM)          // 4096
#define MTOT    (BATCH * LTOT)         // 16384 rows into the projection

typedef __attribute__((ext_vector_type(16))) __bf16 bf16x16;
typedef __attribute__((ext_vector_type(8)))  __bf16 bf16x8;
typedef __attribute__((ext_vector_type(8)))  float  f32x8;

// -------------------------------------------------------------------------
// Kernel 1: horizontal scan.  sequence = (b, w), scan over h.
// One thread per (sequence, channel d). lane -> d gives coalesced x loads.
// Writes y_h as bf16 into ws (full overwrite of poisoned ws).
// -------------------------------------------------------------------------
__global__ void dscan_h(const float* __restrict__ x,
                        const float* __restrict__ A,
                        const float* __restrict__ Bm,
                        const float* __restrict__ Cm,
                        const float* __restrict__ Dsk,
                        __bf16* __restrict__ ysum) {
    const int idx = blockIdx.x * blockDim.x + threadIdx.x;   // 131072 threads
    const int d   = idx & (DMODEL - 1);
    const int seq = idx >> 9;            // b*WDIM + w
    const int b   = seq >> 6;
    const int w   = seq & (WDIM - 1);

    float a[NSTATE], bm[NSTATE], cm[NSTATE], s[NSTATE];
#pragma unroll
    for (int n = 0; n < NSTATE; ++n) {
        a[n]  = A[d * NSTATE + n];
        bm[n] = Bm[d * NSTATE + n];
        cm[n] = Cm[d * NSTATE + n];
        s[n]  = 0.0f;
    }
    const float dsk = Dsk[d];

    const long base   = ((long)b * HDIM * WDIM + w) * DMODEL + d;  // t=0 (h=0)
    const long stride = (long)WDIM * DMODEL;                       // advance h

    for (int t = 0; t < HDIM; ++t) {
        const float xt = x[base + (long)t * stride];
        float y = dsk * xt;
#pragma unroll
        for (int n = 0; n < NSTATE; ++n) {
            s[n] = fmaf(a[n], s[n], bm[n] * xt);
            y    = fmaf(cm[n], s[n], y);
        }
        ysum[base + (long)t * stride] = (__bf16)y;   // y_h lands at (b, h=t, w, d)
    }
}

// -------------------------------------------------------------------------
// Kernel 2: vertical scan.  sequence = (b, h), scan over w.
// Accumulates into ysum:  ysum = bf16( y_h + y_v ).
// -------------------------------------------------------------------------
__global__ void dscan_v(const float* __restrict__ x,
                        const float* __restrict__ A,
                        const float* __restrict__ Bm,
                        const float* __restrict__ Cm,
                        const float* __restrict__ Dsk,
                        __bf16* __restrict__ ysum) {
    const int idx = blockIdx.x * blockDim.x + threadIdx.x;
    const int d   = idx & (DMODEL - 1);
    const int seq = idx >> 9;            // b*HDIM + h
    const int b   = seq >> 6;
    const int h   = seq & (HDIM - 1);

    float a[NSTATE], bm[NSTATE], cm[NSTATE], s[NSTATE];
#pragma unroll
    for (int n = 0; n < NSTATE; ++n) {
        a[n]  = A[d * NSTATE + n];
        bm[n] = Bm[d * NSTATE + n];
        cm[n] = Cm[d * NSTATE + n];
        s[n]  = 0.0f;
    }
    const float dsk = Dsk[d];

    const long base   = ((long)b * HDIM + h) * WDIM * DMODEL + d;  // t=0 (w=0)
    const long stride = (long)DMODEL;                              // advance w

    for (int t = 0; t < WDIM; ++t) {
        const float xt = x[base + (long)t * stride];
        float y = dsk * xt;
#pragma unroll
        for (int n = 0; n < NSTATE; ++n) {
            s[n] = fmaf(a[n], s[n], bm[n] * xt);
            y    = fmaf(cm[n], s[n], y);
        }
        const long p = base + (long)t * stride;
        ysum[p] = (__bf16)((float)ysum[p] + y);      // y_h + y_v
    }
}

// -------------------------------------------------------------------------
// Kernel 3: convert Wp (512x512 f32, row n = output channel) to bf16.
// -------------------------------------------------------------------------
__global__ void cvt_wp(const float* __restrict__ Wp, __bf16* __restrict__ Wpb) {
    const int i = blockIdx.x * blockDim.x + threadIdx.x;   // 262144 threads
    Wpb[i] = (__bf16)Wp[i];
}

// -------------------------------------------------------------------------
// Kernel 4: projection GEMM  out[m][n] = sum_k ysum[m][k] * Wp[n][k] + bias[n]
// via v_wmma_f32_16x16x32_bf16.  One wave owns a 64(M) x 16(N) strip:
//   - B tile (32x16) = 2x8 contiguous K of Wp row (n0+lane%16): two B128 loads
//   - A tile (16x32) = 8 contiguous K of ysum row (m0+lane%16): two B128 loads
//   - 4 f32x8 accumulators reuse B across 4 M-subtiles per K-step.
// Exact grid => EXEC all ones (WMMA requirement).
// -------------------------------------------------------------------------
__global__ void proj_gemm(const __bf16* __restrict__ Ay,   // MTOT x 512 bf16
                          const __bf16* __restrict__ Bw,   // 512  x 512 bf16 (Wp)
                          const float*  __restrict__ bias, // 512  f32
                          float*        __restrict__ out)  // MTOT x 512 f32
{
    const int lane   = threadIdx.x & 31;
    const int waveId = blockIdx.x * (blockDim.x >> 5) + (threadIdx.x >> 5);
    const int tileM  = waveId >> 5;          // 256 strips of 64 rows
    const int tileN  = waveId & 31;          // 32 tiles of 16 cols
    const int m0     = tileM * 64;
    const int n0     = tileN * 16;
    const int row    = lane & 15;            // M (for A) / N (for B/C/D)
    const int half   = lane >> 4;

    f32x8 acc[4] = {};

    const __bf16* brow = Bw + (long)(n0 + row) * DMODEL;

    for (int k0 = 0; k0 < DMODEL; k0 += 32) {
        // B 32x16: lane holds K = half*8 + j (j=0..7) and 16 + half*8 + j of column n0+row
        union { bf16x16 v; bf16x8 h[2]; } bmat;
        bmat.h[0] = *(const bf16x8*)(brow + k0 + half * 8);
        bmat.h[1] = *(const bf16x8*)(brow + k0 + 16 + half * 8);

#pragma unroll
        for (int i = 0; i < 4; ++i) {
            // A 16x32: lane holds K = half*8 + j (j=0..7) and 16 + half*8 + j
            const __bf16* arow = Ay + (long)(m0 + i * 16 + row) * DMODEL + k0;
            union { bf16x16 v; bf16x8 h[2]; } amat;
            amat.h[0] = *(const bf16x8*)(arow + half * 8);
            amat.h[1] = *(const bf16x8*)(arow + 16 + half * 8);

            acc[i] = __builtin_amdgcn_wmma_f32_16x16x32_bf16(
                /*neg_a=*/false, amat.v, /*neg_b=*/false, bmat.v,
                /*c_mod=*/(short)0, acc[i], /*reuse_a=*/false, /*reuse_b=*/false);
        }
    }

    const float bn = bias[n0 + row];
#pragma unroll
    for (int i = 0; i < 4; ++i) {
#pragma unroll
        for (int r = 0; r < 8; ++r) {
            const int m = m0 + i * 16 + half * 8 + r;   // C/D layout: VGPR r -> M
            out[(long)m * DMODEL + (n0 + row)] = acc[i][r] + bn;
        }
    }
}

// -------------------------------------------------------------------------
extern "C" void kernel_launch(void* const* d_in, const int* in_sizes, int n_in,
                              void* d_out, int out_size, void* d_ws, size_t ws_size,
                              hipStream_t stream) {
    const float* x    = (const float*)d_in[0];  // (4, 4096, 512)
    // d_in[1]=h, d_in[2]=w  (scalars, dims hardcoded)
    const float* A    = (const float*)d_in[3];  // (512, 8)
    const float* Bm   = (const float*)d_in[4];  // (512, 8)
    const float* Cm   = (const float*)d_in[5];  // (512, 8)
    const float* Dsk  = (const float*)d_in[6];  // (512,)
    const float* Wp   = (const float*)d_in[7];  // (512, 512)
    const float* bias = (const float*)d_in[8];  // (512,)
    float* out = (float*)d_out;

    // ws layout: [ ysum bf16 : 16 MB ][ Wp bf16 : 512 KB ]
    __bf16* ysum = (__bf16*)d_ws;
    __bf16* wpb  = (__bf16*)((char*)d_ws + (size_t)MTOT * DMODEL * sizeof(__bf16));

    // 131072 threads per scan direction (256 sequences x 512 channels)
    dscan_h<<<512, 256, 0, stream>>>(x, A, Bm, Cm, Dsk, ysum);
    dscan_v<<<512, 256, 0, stream>>>(x, A, Bm, Cm, Dsk, ysum);

    cvt_wp<<<(DMODEL * DMODEL) / 256, 256, 0, stream>>>(Wp, wpb);

    // 8192 waves total: (16384/64 M-strips) x (512/16 N-tiles); 8 waves/block
    proj_gemm<<<8192 / 8, 256, 0, stream>>>(ysum, wpb, bias, out);
}
